// FP8QuantizedLinear_7232724926833
// MI455X (gfx1250) — matrix operand order repaired and checked by reading, not compiled, not run on hardware
//
#include <hip/hip_runtime.h>
#include <hip/hip_bf16.h>
#include <stdint.h>

// ext-vector types matching the gfx1250 WMMA builtin signatures
typedef __attribute__((ext_vector_type(16))) __bf16 v16bf;
typedef __attribute__((ext_vector_type(8)))  float  v8f;
typedef __attribute__((ext_vector_type(4)))  int    v4i;

#define TM    128   // workgroup M tile (8 subtiles of 16 rows)
#define TK    32    // K step per WMMA
#define LDSW  40    // padded LDS row stride in bf16 (80B = 20 banks -> conflict-free)

#ifndef __has_builtin
#define __has_builtin(x) 0
#endif

#if defined(__AMDGCN__) && \
    __has_builtin(__builtin_amdgcn_global_load_async_to_lds_b128) && \
    __has_builtin(__builtin_amdgcn_s_wait_asynccnt)
#define USE_ASYNC_LDS 1
// builtin signature (from hipcc diagnostic): (v4i AS1* global, v4i AS3* lds, imm, imm)
typedef __attribute__((address_space(1))) v4i as1_v4i;
typedef __attribute__((address_space(3))) v4i as3_v4i;
#else
#define USE_ASYNC_LDS 0
#endif

union AFrag { uint32_t u[8]; v16bf v; };

__device__ __forceinline__ void load_afrag(AFrag& f, const __hip_bfloat16* r) {
    *reinterpret_cast<uint4*>(&f.u[0]) = *reinterpret_cast<const uint4*>(r);
    *reinterpret_cast<uint4*>(&f.u[4]) = *reinterpret_cast<const uint4*>(r + 8);
}

__device__ __forceinline__ void dequant_bfrag(v16bf& bfrag, const int4 (&bqv)[4], float s) {
    int q[16];
    q[0]=bqv[0].x; q[1]=bqv[0].y; q[2]=bqv[0].z; q[3]=bqv[0].w;
    q[4]=bqv[1].x; q[5]=bqv[1].y; q[6]=bqv[1].z; q[7]=bqv[1].w;
    q[8]=bqv[2].x; q[9]=bqv[2].y; q[10]=bqv[2].z; q[11]=bqv[2].w;
    q[12]=bqv[3].x; q[13]=bqv[3].y; q[14]=bqv[3].z; q[15]=bqv[3].w;
    #pragma unroll
    for (int e = 0; e < 16; ++e) bfrag[e] = (__bf16)((float)q[e] * s);
}

__device__ __forceinline__ void compute_step(v8f acc[8], const __hip_bfloat16* abase,
                                             const v16bf& bfrag) {
    AFrag af[8];
    #pragma unroll
    for (int st = 0; st < 8; ++st) load_afrag(af[st], abase + st * 16 * LDSW);
    #pragma unroll
    for (int st = 0; st < 8; ++st)
        acc[st] = __builtin_amdgcn_wmma_f32_16x16x32_bf16(
            false, af[st].v, false, bfrag, (short)0, acc[st], false, false);
}

// Split workgroup barrier WITHOUT the implicit loadcnt/dscnt drain of
// __syncthreads(); caller drains exactly the counters that matter (asynccnt).
__device__ __forceinline__ void wg_barrier_raw() {
#if defined(__AMDGCN__)
    asm volatile("s_barrier_signal -1\n\ts_barrier_wait -1" ::: "memory");
#endif
}

// One pipelined K iteration: stage next A tile (async) into `lstage`, fetch next
// B registers into bq_out, dequant+compute current from bq_in / abase, then
// drain ASYNCcnt only and barrier. Buffers/register sets are statically assigned
// by the caller (2x-unrolled alternation) -> no rotation copies, no cndmask.
__device__ __forceinline__ void k_iter(int k0n, int K,
                                       const __hip_bfloat16* __restrict__ aptr,
                                       const int*            __restrict__ wrow,
                                       __hip_bfloat16* lstage,
                                       const __hip_bfloat16* abase,
                                       const int4 (&bq_in)[4], int4 (&bq_out)[4],
                                       v8f acc[8], float s) {
#if USE_ASYNC_LDS
    __builtin_amdgcn_global_load_async_to_lds_b128(
        (as1_v4i*)(aptr + k0n),     (as3_v4i*)lstage,       0, 0);
    __builtin_amdgcn_global_load_async_to_lds_b128(
        (as1_v4i*)(aptr + k0n + 8), (as3_v4i*)(lstage + 8), 0, 0);
#else
    uint4 nac0 = *reinterpret_cast<const uint4*>(aptr + k0n);
    uint4 nac1 = *reinterpret_cast<const uint4*>(aptr + k0n + 8);
#endif
    {
        const int4* p = reinterpret_cast<const int4*>(wrow + k0n);
        bq_out[0] = p[0]; bq_out[1] = p[1]; bq_out[2] = p[2]; bq_out[3] = p[3];
    }
    if (k0n + 3 * TK < K) {
        __builtin_prefetch(wrow + k0n + 3 * TK, 0, 1);
        __builtin_prefetch(aptr + k0n + 3 * TK, 0, 1);
    }

    v16bf bfrag;
    dequant_bfrag(bfrag, bq_in, s);
    compute_step(acc, abase, bfrag);

#if USE_ASYNC_LDS
    __builtin_amdgcn_s_wait_asynccnt(0);
    wg_barrier_raw();
#else
    *reinterpret_cast<uint4*>(lstage)     = nac0;
    *reinterpret_cast<uint4*>(lstage + 8) = nac1;
    __syncthreads();
#endif
}

// out[M,N] = x[M,K] @ dequant(W)[N,K]^T + bias
// Workgroup: 8 waves, 128(M) x 128(N) output block.
// Wave w owns N tile [bx*128 + w*16, +16); all waves share a 128x32 A tile in LDS.
__global__ __launch_bounds__(256)
void fp8q_linear_wmma_bf16(const __hip_bfloat16* __restrict__ x,
                           const int*            __restrict__ qw,
                           const float*          __restrict__ scales,
                           const __hip_bfloat16* __restrict__ bias,
                           __hip_bfloat16*       __restrict__ out,
                           int M, int N, int K) {
    __shared__ __align__(16) __hip_bfloat16 atile[2][TM * LDSW];

    const int t    = threadIdx.x;
    const int lane = t & 31;
    const int wave = t >> 5;
    const int ln   = lane & 15;   // N column (B) / M row (A) within a 16-tile
    const int half = lane >> 4;   // half-wave: selects K sub-range of fragments

    const int mBase = blockIdx.y * TM;
    const int nCol  = blockIdx.x * 128 + wave * 16 + ln;

    const float s     = scales[nCol] * (448.0f / 127.0f);
    const float biasf = (float)bias[nCol];

    // ---- cooperative A-tile staging: thread t copies 32B of row (t>>1) ----
    const int arow = t >> 1;            // 0..127
    const int acol = (t & 1) * 16;      // 0 or 16 (bf16 units)
    const __hip_bfloat16* __restrict__ aptr = x + (size_t)(mBase + arow) * K + acol;
    __hip_bfloat16* const lrow0 = &atile[0][arow * LDSW + acol];
    __hip_bfloat16* const lrow1 = &atile[1][arow * LDSW + acol];

    // ---- B fragment source: W row nCol (one column of B = W^T) ----
    // B k order used by this code (contiguous, same as A): lanes 0-15 hold K=0..15, lanes 16-31 hold K=16..31.
    const int* __restrict__ wrow = qw + (size_t)nCol * K + half * 16;

    // A k order used by this code (contiguous, same as B), per lane: K 0..15 (half=0) / 16..31 (half=1).
    const int aOff = half * 16;
    const __hip_bfloat16* const abase0 = &atile[0][ln * LDSW + aOff];
    const __hip_bfloat16* const abase1 = &atile[1][ln * LDSW + aOff];

    v8f acc[8];
    #pragma unroll
    for (int i = 0; i < 8; ++i) acc[i] = v8f{};

    const int nIter = K / TK;

    // ---------------- prologue: iteration 0 staging + B loads ----------------
    int4 bqA[4], bqB[4];
    {
        const int4* p = reinterpret_cast<const int4*>(wrow);
        bqA[0] = p[0]; bqA[1] = p[1]; bqA[2] = p[2]; bqA[3] = p[3];
    }
#if USE_ASYNC_LDS
    __builtin_amdgcn_global_load_async_to_lds_b128(
        (as1_v4i*)aptr,       (as3_v4i*)lrow0,       0, 0);
    __builtin_amdgcn_global_load_async_to_lds_b128(
        (as1_v4i*)(aptr + 8), (as3_v4i*)(lrow0 + 8), 0, 0);
    __builtin_amdgcn_s_wait_asynccnt(0);
    wg_barrier_raw();
#else
    {
        uint4 ac0 = *reinterpret_cast<const uint4*>(aptr);
        uint4 ac1 = *reinterpret_cast<const uint4*>(aptr + 8);
        *reinterpret_cast<uint4*>(lrow0)     = ac0;
        *reinterpret_cast<uint4*>(lrow0 + 8) = ac1;
    }
    __syncthreads();
#endif

    // ---------------- 2x-unrolled pipelined main loop ----------------
    // Iteration j reads buffer j&1 and B regs (j even: bqA, j odd: bqB),
    // stages buffer (j+1)&1 and the other B register set.
    int i = 0;
    for (; i + 1 < nIter - 1; i += 2) {
        k_iter((i + 1) * TK, K, aptr, wrow, lrow1, abase0, bqA, bqB, acc, s);
        k_iter((i + 2) * TK, K, aptr, wrow, lrow0, abase1, bqB, bqA, acc, s);
    }
    if (i < nIter - 1) {
        // one leftover pipelined iteration (even parity), then final reads buf1/bqB
        k_iter((i + 1) * TK, K, aptr, wrow, lrow1, abase0, bqA, bqB, acc, s);
        v16bf bfrag;
        dequant_bfrag(bfrag, bqB, s);
        compute_step(acc, abase1, bfrag);
    } else {
        // final iteration has even parity: reads buf0/bqA
        v16bf bfrag;
        dequant_bfrag(bfrag, bqA, s);
        compute_step(acc, abase0, bfrag);
    }

    // ---- epilogue: bias add + bf16 store per C/D layout ----
    // VGPR r: lanes 0-15 -> M = r, lanes 16-31 -> M = r + 8; N = lane & 15.
    #pragma unroll
    for (int stile = 0; stile < 8; ++stile) {
        const int mrow = mBase + stile * 16 + half * 8;
        #pragma unroll
        for (int r = 0; r < 8; ++r) {
            float v = acc[stile][r] + biasf;
            out[(size_t)(mrow + r) * N + nCol] = (__hip_bfloat16)v;
        }
    }
}

extern "C" void kernel_launch(void* const* d_in, const int* in_sizes, int n_in,
                              void* d_out, int out_size, void* d_ws, size_t ws_size,
                              hipStream_t stream) {
    const __hip_bfloat16* x      = (const __hip_bfloat16*)d_in[0];
    const int*            qw     = (const int*)d_in[1];
    const float*          scales = (const float*)d_in[2];
    const __hip_bfloat16* bias   = (const __hip_bfloat16*)d_in[3];
    __hip_bfloat16*       out    = (__hip_bfloat16*)d_out;

    const int N = in_sizes[3];            // OUT_F (bias length)
    const int K = in_sizes[1] / N;        // IN_F
    const int M = in_sizes[0] / K;        // B * S

    dim3 block(256);
    dim3 grid((unsigned)(N / 128), (unsigned)(M / TM));
    hipLaunchKernelGGL(fp8q_linear_wmma_bf16, grid, block, 0, stream,
                       x, qw, scales, bias, out, M, N, K);
}
